// GraphOnlyMasifSiteWrapper_51908974739857
// MI455X (gfx1250) — matrix operand order repaired and checked by reading, not compiled, not run on hardware
//
#include <hip/hip_runtime.h>

// Problem constants (fixed by the reference harness)
#define BATCH 16
#define NS    4096
#define NG    2048
#define DIM   128
#define HID   64
#define TILES_PER_WAVE 4   // 128 blocks * 8 waves * 4 tiles * 16 rows = 65536

typedef __attribute__((ext_vector_type(16))) _Float16 v16h;
typedef __attribute__((ext_vector_type(8)))  float    v8f;

// ---------------------------------------------------------------------------
// Kernel 1: per-protein nearest-neighbor argmin.
// One block = 256 surface vertices of one protein. Graph positions + |g|^2
// staged in LDS as float4 -> inner loop is a ds_load_b128 broadcast + 3 FMAs.
// score = |g|^2 - 2 s.g  (same argmin ordering as squared distance).
// ---------------------------------------------------------------------------
__global__ __launch_bounds__(256) void nn_kernel(const float* __restrict__ gpos,
                                                 const float* __restrict__ spos,
                                                 int* __restrict__ nn_idx) {
    __shared__ float4 gp[NG];                 // x, y, z, |g|^2  (32 KB)
    const int batch = blockIdx.x >> 4;        // 16 blocks per protein
    const float* gbase = gpos + (size_t)batch * NG * 3;
    for (int i = threadIdx.x; i < NG; i += 256) {
        float gx = gbase[3 * i + 0];
        float gy = gbase[3 * i + 1];
        float gz = gbase[3 * i + 2];
        float g2 = fmaf(gx, gx, fmaf(gy, gy, gz * gz));
        gp[i] = make_float4(gx, gy, gz, g2);
    }
    __syncthreads();

    const int v  = blockIdx.x * 256 + threadIdx.x;    // global surface vertex
    const float nsx = -2.0f * spos[3 * v + 0];
    const float nsy = -2.0f * spos[3 * v + 1];
    const float nsz = -2.0f * spos[3 * v + 2];

    float best = 3.4e38f;
    int   bi   = 0;
    #pragma unroll 8
    for (int j = 0; j < NG; ++j) {
        float4 g = gp[j];                              // broadcast read
        float s = fmaf(nsx, g.x, g.w);
        s = fmaf(nsy, g.y, s);
        s = fmaf(nsz, g.z, s);
        bool lt = s < best;                            // strict: keep first min
        best = lt ? s : best;
        bi   = lt ? j : bi;
    }
    nn_idx[v] = batch * NG + bi;                       // global graph-node index
}

// ---------------------------------------------------------------------------
// Kernel 2: fused gather + MLP using v_wmma_f32_16x16x32_f16.
// 128 blocks x 256 threads (8 waves). Each wave processes 4 consecutive
// 16-row tiles. W1 is swizzled once per block into WMMA-B fragment layout in
// LDS, then hoisted into registers (bfr[16], 128 VGPRs) so the steady-state
// loop has ZERO LDS traffic: global b128 gather + cvt (co-executes with XDL
// WMMAs) feeding 16 back-to-back v_wmma ops per tile.
// Layer 2 (H x 1) is folded into the C/D lane layout + shfl_xor reduction.
// ---------------------------------------------------------------------------
__global__ __launch_bounds__(256) void mlp_kernel(const float* __restrict__ gf,
                                                  const int* __restrict__ nn_idx,
                                                  const float* __restrict__ W1,
                                                  const float* __restrict__ b1,
                                                  const float* __restrict__ W2,
                                                  const float* __restrict__ b2,
                                                  float* __restrict__ out) {
    // 16 fragments (4 N-tiles x 4 K-steps) x 32 lanes x 16 halfs = 16 KB.
    __shared__ v16h  w1frag[16 * 32];
    __shared__ float b1s[HID];
    __shared__ float w2s[HID];

    // Stage + swizzle W1 into fragment-major layout. Slot (frag, lane) covers
    // B elements with t = frag>>2, s = frag&3, m = lane&15, hh = lane>>4,
// element i -> K = 32*s + 8*hh + (i<8 ? i : i+8), N = 16*t + m.
    for (int e = threadIdx.x; e < 16 * 32; e += 256) {
        const int frag = e >> 5;
        const int ln   = e & 31;
        const int t    = frag >> 2;
        const int s    = frag & 3;
        const int kb   = 32 * s + 8 * (ln >> 4);
        const int n    = 16 * t + (ln & 15);
        v16h fragv;
        #pragma unroll
        for (int i = 0; i < 16; ++i)
            fragv[i] = (_Float16)W1[(kb + (i < 8 ? i : i + 8)) * HID + n];
        w1frag[e] = fragv;
    }
    if (threadIdx.x < HID) {
        b1s[threadIdx.x] = b1[threadIdx.x];
        w2s[threadIdx.x] = W2[threadIdx.x];
    }
    __syncthreads();

    const int wave = threadIdx.x >> 5;
    const int lane = threadIdx.x & 31;
    const int m    = lane & 15;           // N-position / A-row within tile
    const int h    = lane >> 4;           // lane half selects K sub-range

    // Hoist all 16 B fragments into registers (loop-invariant operands).
    v16h bfr[16];
    #pragma unroll
    for (int q = 0; q < 16; ++q)
        bfr[q] = w1frag[q * 32 + lane];

    // Per-lane loop-invariant layer-2 constants for N = 16*t + m.
    float w2v[4], b1v[4];
    #pragma unroll
    for (int t = 0; t < 4; ++t) {
        w2v[t] = w2s[16 * t + m];
        b1v[t] = b1s[16 * t + m];
    }
    const float bb = b2[0];

    const int wavebase = (blockIdx.x * 8 + wave) * (TILES_PER_WAVE * 16);

    for (int it = 0; it < TILES_PER_WAVE; ++it) {
        const int rowbase = wavebase + it * 16;
        const int row     = rowbase + m;               // this lane's A row
        const float* f = gf + (size_t)nn_idx[row] * DIM;

        // Prefetch next tile's gathered row (global_prefetch_b8).
        if (it + 1 < TILES_PER_WAVE)
            __builtin_prefetch(gf + (size_t)nn_idx[row + 16] * DIM, 0, 3);

        // A fragments, 4 K-steps of 32. ISA layout: element i -> K =
        // 32*s + 8*h + (i<8 ? i : i+8). Aligned 8-float runs -> b128 clauses.
        v16h a[4];
        #pragma unroll
        for (int s = 0; s < 4; ++s) {
            const int k0 = 32 * s + 8 * h;
            #pragma unroll
            for (int i = 0; i < 8; ++i) {
                a[s][i]     = (_Float16)f[k0 + i];
                a[s][8 + i] = (_Float16)f[k0 + 16 + i];
            }
        }

        float outp[8];
        #pragma unroll
        for (int r = 0; r < 8; ++r) outp[r] = 0.0f;

        #pragma unroll
        for (int t = 0; t < 4; ++t) {                  // 4 N-tiles cover H=64
            v8f c = {};
            #pragma unroll
            for (int s = 0; s < 4; ++s)                // 4 K-steps cover D=128
                c = __builtin_amdgcn_wmma_f32_16x16x32_f16(
                        false, a[s], false, bfr[t * 4 + s], (short)0, c,
                        false, false);
            // Fold layer 2: this lane's C elements sit at N = 16*t + m.
            #pragma unroll
            for (int r = 0; r < 8; ++r) {
                float hv = c[r] + b1v[t];
                hv = hv > 0.0f ? hv : 0.0f;            // relu
                outp[r] = fmaf(hv, w2v[t], outp[r]);
            }
        }

        // Reduce over N: sum across the 16 lanes of each half (xor 1,2,4,8
        // never crosses the 16-lane boundary in wave32).
        #pragma unroll
        for (int off = 1; off < 16; off <<= 1) {
            #pragma unroll
            for (int r = 0; r < 8; ++r)
                outp[r] += __shfl_xor(outp[r], off, 32);
        }

        if (m == 0) {
            #pragma unroll
            for (int r = 0; r < 8; ++r)                // M = r + 8*h
                out[rowbase + 8 * h + r] = outp[r] + bb;
        }
    }
}

// ---------------------------------------------------------------------------
// Inputs (setup_inputs order):
// 0 graph_features [B*NG, D] f32   1 graph_pos [B*NG, 3] f32
// 2 surface_pos [B*NS, 3] f32      3 surface_batch (unused)
// 4 graph_batch (unused)           5 num_batches (unused, fixed 16)
// 6 W1 [D,H] f32  7 b1 [H] f32  8 W2 [H,1] f32  9 b2 [1] f32
// d_out: [B*NS] f32. d_ws: nn indices (B*NS ints = 256 KB).
// ---------------------------------------------------------------------------
extern "C" void kernel_launch(void* const* d_in, const int* in_sizes, int n_in,
                              void* d_out, int out_size, void* d_ws, size_t ws_size,
                              hipStream_t stream) {
    const float* gf   = (const float*)d_in[0];
    const float* gpos = (const float*)d_in[1];
    const float* spos = (const float*)d_in[2];
    const float* W1   = (const float*)d_in[6];
    const float* b1   = (const float*)d_in[7];
    const float* W2   = (const float*)d_in[8];
    const float* b2   = (const float*)d_in[9];
    int* nn = (int*)d_ws;

    nn_kernel<<<(BATCH * NS) / 256, 256, 0, stream>>>(gpos, spos, nn);
    mlp_kernel<<<(BATCH * NS) / (8 * TILES_PER_WAVE * 16), 256, 0, stream>>>(
        gf, nn, W1, b1, W2, b2, (float*)d_out);
}
